// CausalSelfAttention_7232724926954
// MI455X (gfx1250) — compile-verified, matrix operand order checked by reading, not run on hardware
//
#include <hip/hip_runtime.h>

// ---------------- problem constants ----------------
#define BB    2
#define SS    1024
#define CC    1024
#define NTOK  2048          // BB*SS
#define HH    16
#define DD    64
#define RR    16
#define RKK   32
#define FAA   32

// ---------------- CDNA5 WMMA types ----------------
typedef __attribute__((ext_vector_type(16))) __bf16 v16bf;
typedef __attribute__((ext_vector_type(8)))  float  v8f;
typedef __attribute__((ext_vector_type(4)))  unsigned int v4u;
typedef __attribute__((ext_vector_type(8)))  int    v8i;
typedef __attribute__((ext_vector_type(4)))  int    v4i;

__device__ __forceinline__ __bf16 f2bf(float f) {
  union { float f; unsigned u; } x; x.f = f;
  unsigned r = (x.u + 0x7fffu + ((x.u >> 16) & 1u)) >> 16;
  unsigned short h = (unsigned short)r;
  __bf16 o;
  __builtin_memcpy(&o, &h, 2);
  return o;
}

__device__ __forceinline__ v8f wmma_bf16(v16bf a, v16bf b, v8f c) {
  return __builtin_amdgcn_wmma_f32_16x16x32_bf16(false, a, false, b, (short)0, c,
                                                 false, false);
}

// K index pattern of 16-bit A/B fragments (wave32):
//   lanes 0-15 : elems 0..15 -> K {0..7, 16..23}
//   lanes16-31 : elems 0..15 -> K {8..15, 24..31}
// frag_k: element index -> K   (used for scattered global gathers)
__device__ __forceinline__ int frag_k(int i, int lane) {
  return (i < 8 ? i : i + 8) + ((lane >> 4) << 3);
}
// fragpos: K -> storage position such that a lane's 16 elems are contiguous.
// (swap bits 3 and 4 of k)
__device__ __forceinline__ int fragpos(int k) {
  return (k & 7) | ((k >> 1) & 8) | ((k << 1) & 16);
}

// ---------------- Tensor Data Mover: tile_d1 x 32 fp32 tile -> LDS ----------
__device__ __forceinline__ void tdm_load_tile_f32(void* lds_ptr,
                                                  const float* gptr,
                                                  int row_stride_elems,
                                                  int tile_d0, int tile_d1) {
  unsigned lds = (unsigned)(unsigned long long)(uintptr_t)lds_ptr;
  unsigned long long ga = (unsigned long long)(uintptr_t)gptr;
  v4u g0 = { 1u,                                   // count=1, user mode
             lds,                                  // lds_addr  [63:32]
             (unsigned)(ga & 0xffffffffull),       // global_addr lo
             (unsigned)((ga >> 32) & 0x1ffffffull) | (2u << 30) }; // hi + type=2
  v8i g1 = { (int)(2u << 16),                      // mask=0, data_size=2 (4B)
             (int)((unsigned)tile_d0 << 16),       // tensor_dim0 lo16 @ bit48
             (int)((unsigned)tile_d1 << 16),       // dim0 hi=0 | tensor_dim1 lo16
             (int)((unsigned)tile_d0 << 16),       // dim1 hi=0 | tile_dim0
             (int)tile_d1,                         // tile_dim1, tile_dim2=0
             (int)row_stride_elems,                // tensor_dim0_stride lo32
             0, 0 };
  v4i g2 = {0, 0, 0, 0};
  v4i g3 = {0, 0, 0, 0};
#if defined(__clang_major__) && (__clang_major__ >= 23)
  v8i g4 = {0, 0, 0, 0, 0, 0, 0, 0};
  __builtin_amdgcn_tensor_load_to_lds(g0, g1, g2, g3, g4, 0);
#else
  __builtin_amdgcn_tensor_load_to_lds(g0, g1, g2, g3, 0);
#endif
}

// ------- generic GEMM: C = A(MxK) * B(KxN); block tile 64x128, wave 32x32 ----
__global__ __launch_bounds__(256) void gemm_kernel(const float* __restrict__ A,
                                                   const float* __restrict__ B,
                                                   float* __restrict__ C,
                                                   int M, int N, int K) {
  __shared__ float  Ast[64 * 32];      // TDM staging (raw fp32)
  __shared__ __bf16 Asf[64][48];       // fragment-order bf16
  __shared__ __bf16 Bsf[128][48];      // fragment-order bf16 (transposed panel)
  const int bm   = blockIdx.x * 64;
  const int bn   = blockIdx.y * 128;
  const int wave = threadIdx.x >> 5;
  const int lane = threadIdx.x & 31;
  const int wr   = (wave & 1) * 32;
  const int wc   = (wave >> 1) * 32;
  const int mrow = lane & 15;
  const int koff = (lane >> 4) << 4;   // 0 or 16 -> contiguous 32B frag slice
  v8f acc[2][2] = {};
  for (int k0 = 0; k0 < K; k0 += 32) {
    if (wave == 0) {
      // async tensor DMA of the 64x32 A tile into LDS staging
      tdm_load_tile_f32(Ast, A + (size_t)bm * K + k0, K, 32, 64);
    }
    if (threadIdx.x == 64 && k0 + 32 < K)
      __builtin_prefetch(A + (size_t)bm * K + k0 + 32, 0, 1);
    // B panel 32x128 -> frag-order bf16 LDS (bounds-checked on N)
    for (int i = threadIdx.x; i < 32 * 128; i += 256) {
      int r = i >> 7, c = i & 127;
      int col = bn + c;
      float v = (col < N) ? B[(size_t)(k0 + r) * N + col] : 0.f;
      Bsf[c][fragpos(r)] = f2bf(v);
    }
    if (wave == 0) __builtin_amdgcn_s_wait_tensorcnt(0);
    __syncthreads();
    for (int i = threadIdx.x; i < 64 * 32; i += 256)
      Asf[i >> 5][fragpos(i & 31)] = f2bf(Ast[i]);
    __syncthreads();
    v16bf af[2], bf[2];
#pragma unroll
    for (int t = 0; t < 2; ++t) {
      af[t] = *(const v16bf*)&Asf[wr + t * 16 + mrow][koff];
      bf[t] = *(const v16bf*)&Bsf[wc + t * 16 + mrow][koff];
    }
#pragma unroll
    for (int rt = 0; rt < 2; ++rt)
#pragma unroll
      for (int ct = 0; ct < 2; ++ct)
        acc[rt][ct] = wmma_bf16(af[rt], bf[ct], acc[rt][ct]);
    __syncthreads();
  }
  const int roff = (lane >= 16) ? 8 : 0;
#pragma unroll
  for (int rt = 0; rt < 2; ++rt)
#pragma unroll
    for (int ct = 0; ct < 2; ++ct) {
      int col = bn + wc + ct * 16 + mrow;
      if (col < N) {
        int r0 = bm + wr + rt * 16 + roff;
#pragma unroll
        for (int j = 0; j < 8; ++j)
          C[(size_t)(r0 + j) * N + col] = acc[rt][ct][j];
      }
    }
}

// ------------- per-token rule term: dst += (V[rid] @ (xm @ U[rid]^T)) * g ----
__global__ __launch_bounds__(256) void rule_term_kernel(
    const float* __restrict__ src, const float* __restrict__ rU,
    const float* __restrict__ rV, const float* __restrict__ gain,
    const int* __restrict__ rids, float* __restrict__ dst) {
  __shared__ __bf16 xusf[8][32][48];   // xu stored as [c][fragpos(b)]
  const int wave = threadIdx.x >> 5;
  const int lane = threadIdx.x & 31;
  const int mrow = lane & 15;
  const int roff = (lane >= 16) ? 8 : 0;
  const int koff = (lane >> 4) << 4;
  const int n    = blockIdx.x * 8 + wave;
  const int rid  = rids[n];
  const float* xrow = src + (size_t)n * CC;
  const float* U = rU + (size_t)rid * (FAA * FAA);
  const float* V = rV + (size_t)rid * (FAA * FAA);
  const float g = gain[rid];

  // xu[b][c] = sum_a xm[b][a] * U[c][a]   (A = xm, B = U^T)
  v16bf ax[2];
#pragma unroll
  for (int bt = 0; bt < 2; ++bt)
#pragma unroll
    for (int i = 0; i < 16; ++i)
      ax[bt][i] = f2bf(xrow[(bt * 16 + mrow) * 32 + frag_k(i, lane)]);
#pragma unroll
  for (int ct = 0; ct < 2; ++ct) {
    v16bf bu;
#pragma unroll
    for (int i = 0; i < 16; ++i)
      bu[i] = f2bf(U[(ct * 16 + mrow) * 32 + frag_k(i, lane)]);
#pragma unroll
    for (int bt = 0; bt < 2; ++bt) {
      v8f acc = {};
      acc = wmma_bf16(ax[bt], bu, acc);
#pragma unroll
      for (int j = 0; j < 8; ++j)
        xusf[wave][ct * 16 + mrow][fragpos(bt * 16 + roff + j)] = f2bf(acc[j]);
    }
  }
  __syncthreads();
  // vxu[d][c] = sum_b V[d][b] * xu[b][c]
  v16bf av[2], bx[2];
#pragma unroll
  for (int t = 0; t < 2; ++t) {
#pragma unroll
    for (int i = 0; i < 16; ++i)
      av[t][i] = f2bf(V[(t * 16 + mrow) * 32 + frag_k(i, lane)]);
    bx[t] = *(const v16bf*)&xusf[wave][t * 16 + mrow][koff];
  }
#pragma unroll
  for (int dt = 0; dt < 2; ++dt)
#pragma unroll
    for (int ct = 0; ct < 2; ++ct) {
      v8f acc = {};
      acc = wmma_bf16(av[dt], bx[ct], acc);
#pragma unroll
      for (int j = 0; j < 8; ++j) {
        int d = dt * 16 + roff + j;
        int c = ct * 16 + mrow;
        dst[(size_t)n * CC + d * 32 + c] += acc[j] * g;
      }
    }
}

// ---------------- RoPE (in place, layout (n, h*64+d)) ----------------
__global__ void rope_kernel(float* __restrict__ t) {
  int i = blockIdx.x * blockDim.x + threadIdx.x;  // pair index over NTOK*512
  if (i >= NTOK * (CC / 2)) return;
  int n  = i >> 9;           // token
  int pr = i & 511;          // h*32 + d2
  int d2 = pr & 31;
  int s  = n & (SS - 1);
  float freq = __expf((float)(2 * d2) * (-9.210340371976184f / 64.f));
  float sn, cs;
  __sincosf((float)s * freq, &sn, &cs);
  size_t base = (size_t)n * CC + (size_t)(pr >> 5) * 64 + (size_t)d2 * 2;
  float t1 = t[base], t2 = t[base + 1];
  t[base]     = t1 * cs - t2 * sn;
  t[base + 1] = t2 * cs + t1 * sn;
}

// ---------------- flash attention, causal, WMMA QK^T and PV ----------------
__global__ __launch_bounds__(256) void flash_attn_kernel(
    const float* __restrict__ qb, const float* __restrict__ kb,
    const float* __restrict__ vb, float* __restrict__ ctx) {
  __shared__ __bf16 Kf[32][80];        // [key][slice*32 + fragpos(d&31)]
  __shared__ __bf16 Vf[64][48];        // [d][fragpos(key)]
  __shared__ __bf16 psf[8][16][48];    // per-wave P, fragment order
  const int wave = threadIdx.x >> 5;
  const int lane = threadIdx.x & 31;
  const int mrow = lane & 15;
  const int roff = (lane >= 16) ? 8 : 0;
  const int koff = (lane >> 4) << 4;
  const int bh = blockIdx.y;
  const int b = bh >> 4, h = bh & 15;
  const int q0 = blockIdx.x * 128 + wave * 16;

  // Q fragments (two K slices of 32 over D=64)
  v16bf qa[2];
#pragma unroll
  for (int s2 = 0; s2 < 2; ++s2)
#pragma unroll
    for (int i = 0; i < 16; ++i) {
      int d = s2 * 32 + frag_k(i, lane);
      qa[s2][i] = f2bf(qb[(size_t)(b * SS + q0 + mrow) * CC + h * 64 + d]);
    }

  float m_r[8], l_r[8];
  v8f o[4] = {};
#pragma unroll
  for (int j = 0; j < 8; ++j) { m_r[j] = -1e30f; l_r[j] = 0.f; }

  const int ks_hi = (blockIdx.x + 1) * 128;   // block-uniform bound
  for (int ks = 0; ks < ks_hi; ks += 32) {
    // cooperative stage of K and V chunk (32 keys x 64 dims) into frag-order LDS
    for (int i = threadIdx.x; i < 32 * 64; i += 256) {
      int key = i >> 6, d = i & 63;
      int kg = ks + key; if (kg > SS - 1) kg = SS - 1;   // masked later
      size_t base = (size_t)(b * SS + kg) * CC + h * 64 + d;
      Kf[key][(d & 32) + fragpos(d & 31)] = f2bf(kb[base]);
      Vf[d][fragpos(key)] = f2bf(vb[base]);
    }
    __syncthreads();
    if (ks < q0 + 16) {                 // wave-uniform: EXEC all-ones inside
      // scores: 16 queries x 32 keys
      v8f sc[2] = {};
#pragma unroll
      for (int nt = 0; nt < 2; ++nt)
#pragma unroll
        for (int s2 = 0; s2 < 2; ++s2) {
          v16bf bk = *(const v16bf*)&Kf[nt * 16 + mrow][s2 * 32 + koff];
          sc[nt] = wmma_bf16(qa[s2], bk, sc[nt]);
        }
      // scale + causal mask + online softmax (rows live in half-waves)
      float mloc[8], alpha[8];
#pragma unroll
      for (int j = 0; j < 8; ++j) {
        int row = q0 + roff + j;
        float v0 = sc[0][j] * 0.125f, v1 = sc[1][j] * 0.125f;
        if (ks + 0 * 16 + mrow > row) v0 = -1e30f;
        if (ks + 1 * 16 + mrow > row) v1 = -1e30f;
        sc[0][j] = v0; sc[1][j] = v1;
        float m = fmaxf(v0, v1);
        m = fmaxf(m, __shfl_xor(m, 1, 32));
        m = fmaxf(m, __shfl_xor(m, 2, 32));
        m = fmaxf(m, __shfl_xor(m, 4, 32));
        m = fmaxf(m, __shfl_xor(m, 8, 32));
        mloc[j] = m;
      }
#pragma unroll
      for (int j = 0; j < 8; ++j) {
        float mn = fmaxf(m_r[j], mloc[j]);
        alpha[j] = __expf(m_r[j] - mn);
        float p0 = __expf(sc[0][j] - mn);
        float p1 = __expf(sc[1][j] - mn);
        sc[0][j] = p0; sc[1][j] = p1;
        float rs = p0 + p1;
        rs += __shfl_xor(rs, 1, 32);
        rs += __shfl_xor(rs, 2, 32);
        rs += __shfl_xor(rs, 4, 32);
        rs += __shfl_xor(rs, 8, 32);
        m_r[j] = mn;
        l_r[j] = l_r[j] * alpha[j] + rs;
      }
#pragma unroll
      for (int t = 0; t < 4; ++t)
#pragma unroll
        for (int j = 0; j < 8; ++j) o[t][j] *= alpha[j];
      // stage P in fragment order for the PV matmul
#pragma unroll
      for (int nt = 0; nt < 2; ++nt)
#pragma unroll
        for (int j = 0; j < 8; ++j)
          psf[wave][roff + j][fragpos(nt * 16 + mrow)] = f2bf(sc[nt][j]);
      asm volatile("s_wait_dscnt 0x0" ::: "memory");  // wave-local LDS RAW fence
      v16bf pa = *(const v16bf*)&psf[wave][mrow][koff];
#pragma unroll
      for (int dt = 0; dt < 4; ++dt) {
        v16bf bv = *(const v16bf*)&Vf[dt * 16 + mrow][koff];
        o[dt] = wmma_bf16(pa, bv, o[dt]);
      }
    }
    __syncthreads();
  }
  // normalize + store ctx
#pragma unroll
  for (int j = 0; j < 8; ++j) {
    float inv = 1.f / l_r[j];
    int row = q0 + roff + j;
#pragma unroll
    for (int dt = 0; dt < 4; ++dt)
      ctx[(size_t)(b * SS + row) * CC + h * 64 + dt * 16 + mrow] = o[dt][j] * inv;
  }
}

// ---------------- hier path ----------------
__global__ void zero_kernel(float* __restrict__ p, int n) {
  int i = blockIdx.x * blockDim.x + threadIdx.x;
  if (i < n) p[i] = 0.f;
}

__global__ __launch_bounds__(64) void hier_logits_kernel(
    const float* __restrict__ qh, const float* __restrict__ kv,
    const int* __restrict__ rids, float* __restrict__ logits) {
  __shared__ float red[16];
  int b = blockIdx.x;
  int c = blockIdx.y * 64 + threadIdx.x;
  int lane = threadIdx.x & 31;
  float ksum[16]; int cnt[16];
#pragma unroll
  for (int u = 0; u < 16; ++u) { ksum[u] = 0.f; cnt[u] = 0; }
  for (int s = 0; s < SS; ++s) {
    int row = b * SS + s;
    int rid = rids[row];
    ksum[rid] += kv[(size_t)row * (2 * CC) + c];
    cnt[rid] += 1;
    float qv = qh[(size_t)row * CC + c];
    if (threadIdx.x < 16) red[threadIdx.x] = 0.f;
    __syncthreads();
    for (int u = 0; u < 16; ++u) {
      int cc = cnt[u] > 0 ? cnt[u] : 1;
      float part = qv * ksum[u] / (float)cc;
      part += __shfl_xor(part, 16, 32);
      part += __shfl_xor(part, 8, 32);
      part += __shfl_xor(part, 4, 32);
      part += __shfl_xor(part, 2, 32);
      part += __shfl_xor(part, 1, 32);
      if (lane == 0) atomicAdd(&red[u], part);
    }
    __syncthreads();
    if (threadIdx.x < 16) atomicAdd(&logits[row * 16 + threadIdx.x], red[threadIdx.x]);
    __syncthreads();
  }
}

__global__ void hier_softmax_kernel(const float* __restrict__ logits,
                                    float* __restrict__ w) {
  int i = blockIdx.x * blockDim.x + threadIdx.x;
  if (i >= NTOK) return;
  float v[16], mx = -1e30f;
#pragma unroll
  for (int u = 0; u < 16; ++u) { v[u] = logits[i * 16 + u] * 0.03125f; mx = fmaxf(mx, v[u]); }
  float sum = 0.f;
#pragma unroll
  for (int u = 0; u < 16; ++u) { v[u] = __expf(v[u] - mx); sum += v[u]; }
  float inv = 1.f / sum;
#pragma unroll
  for (int u = 0; u < 16; ++u) w[i * 16 + u] = v[u] * inv;
}

__global__ __launch_bounds__(64) void hier_out_kernel(
    const float* __restrict__ kv, const int* __restrict__ rids,
    const float* __restrict__ w, const float* __restrict__ gate,
    float* __restrict__ out) {
  int b = blockIdx.x;
  int c = blockIdx.y * 64 + threadIdx.x;
  float vsum[16]; int cnt[16];
#pragma unroll
  for (int u = 0; u < 16; ++u) { vsum[u] = 0.f; cnt[u] = 0; }
  float gv = gate[c];
  for (int s = 0; s < SS; ++s) {
    int row = b * SS + s;
    int rid = rids[row];
    vsum[rid] += kv[(size_t)row * (2 * CC) + CC + c];
    cnt[rid] += 1;
    float acc = 0.f;
    for (int u = 0; u < 16; ++u) {
      int cc = cnt[u] > 0 ? cnt[u] : 1;
      acc += w[row * 16 + u] * vsum[u] / (float)cc;
    }
    out[(size_t)row * CC + c] += acc * gv;
  }
}

// ---------------- host launcher ----------------
extern "C" void kernel_launch(void* const* d_in, const int* in_sizes, int n_in,
                              void* d_out, int out_size, void* d_ws, size_t ws_size,
                              hipStream_t stream) {
  (void)in_sizes; (void)n_in; (void)out_size; (void)ws_size;
  const float* x          = (const float*)d_in[0];
  const float* shared_in  = (const float*)d_in[1];   // 4 x C x RK
  const float* shared_out = (const float*)d_in[2];   // 4 x RK x C
  const float* rule_U     = (const float*)d_in[3];   // 4 x R x 32 x 32
  const float* rule_V     = (const float*)d_in[4];
  const float* rule_gain  = (const float*)d_in[5];   // 4 x R
  const float* wq         = (const float*)d_in[6];
  const float* wkv        = (const float*)d_in[7];
  const float* gate       = (const float*)d_in[8];
  const int*   rule_ids   = (const int*)d_in[9];
  float* out = (float*)d_out;

  float* ws = (float*)d_ws;
  float* qbuf   = ws;                       // NTOK*C
  float* kbuf   = qbuf + (size_t)NTOK * CC;
  float* vbuf   = kbuf + (size_t)NTOK * CC;
  float* ctx    = vbuf + (size_t)NTOK * CC;
  float* tbuf   = ctx  + (size_t)NTOK * CC; // NTOK*RK
  float* qh     = tbuf + (size_t)NTOK * RKK;
  float* kv     = qh   + (size_t)NTOK * CC; // NTOK*2C
  float* logits = kv   + (size_t)NTOK * 2 * CC;
  float* wbuf   = logits + (size_t)NTOK * RR;

  // q,k,v projections: x @ Si @ So + rule term
  float* dsts[3] = {qbuf, kbuf, vbuf};
  for (int p = 0; p < 3; ++p) {
    gemm_kernel<<<dim3(NTOK / 64, 1), 256, 0, stream>>>(
        x, shared_in + (size_t)p * CC * RKK, tbuf, NTOK, RKK, CC);
    gemm_kernel<<<dim3(NTOK / 64, CC / 128), 256, 0, stream>>>(
        tbuf, shared_out + (size_t)p * RKK * CC, dsts[p], NTOK, CC, RKK);
    rule_term_kernel<<<NTOK / 8, 256, 0, stream>>>(
        x, rule_U + (size_t)p * RR * FAA * FAA, rule_V + (size_t)p * RR * FAA * FAA,
        rule_gain + (size_t)p * RR, rule_ids, dsts[p]);
  }
  // RoPE on q and k
  int rope_threads = NTOK * (CC / 2);
  rope_kernel<<<(rope_threads + 255) / 256, 256, 0, stream>>>(qbuf);
  rope_kernel<<<(rope_threads + 255) / 256, 256, 0, stream>>>(kbuf);
  // causal attention
  flash_attn_kernel<<<dim3(SS / 128, BB * HH), 256, 0, stream>>>(qbuf, kbuf, vbuf, ctx);
  // output projection
  gemm_kernel<<<dim3(NTOK / 64, 1), 256, 0, stream>>>(
      ctx, shared_in + (size_t)3 * CC * RKK, tbuf, NTOK, RKK, CC);
  gemm_kernel<<<dim3(NTOK / 64, CC / 128), 256, 0, stream>>>(
      tbuf, shared_out + (size_t)3 * RKK * CC, out, NTOK, CC, RKK);
  rule_term_kernel<<<NTOK / 8, 256, 0, stream>>>(
      ctx, rule_U + (size_t)3 * RR * FAA * FAA, rule_V + (size_t)3 * RR * FAA * FAA,
      rule_gain + (size_t)3 * RR, rule_ids, out);
  // hier path
  gemm_kernel<<<dim3(NTOK / 64, CC / 128), 256, 0, stream>>>(x, wq, qh, NTOK, CC, CC);
  gemm_kernel<<<dim3(NTOK / 64, (2 * CC) / 128), 256, 0, stream>>>(x, wkv, kv, NTOK, 2 * CC, CC);
  zero_kernel<<<(NTOK * RR + 255) / 256, 256, 0, stream>>>(logits, NTOK * RR);
  hier_logits_kernel<<<dim3(BB, CC / 64), 64, 0, stream>>>(qh, kv, rule_ids, logits);
  hier_softmax_kernel<<<(NTOK + 255) / 256, 256, 0, stream>>>(logits, wbuf);
  hier_out_kernel<<<dim3(BB, CC / 64), 64, 0, stream>>>(kv, rule_ids, wbuf, gate, out);
}
